// ProjectionLayer_65833258713767
// MI455X (gfx1250) — matrix operand order repaired and checked by reading, not compiled, not run on hardware
//
#include <hip/hip_runtime.h>
#include <hip/hip_bf16.h>
#include <math.h>

// Problem constants from the reference (fixed harness problem):
#define B_DIM 8
#define S_DIM 4096
#define H_DIM 1024
#define M_DIM 256
#define R_DIM (B_DIM * M_DIM)        // 2048 compacted rows for the GEMM
#define LOG2_S 12                    // S_DIM == 4096

typedef __attribute__((ext_vector_type(16))) __bf16 v16bf;
typedef __attribute__((ext_vector_type(8)))  __bf16 v8bf;
typedef __attribute__((ext_vector_type(4)))  __bf16 v4bf;
typedef __attribute__((ext_vector_type(8)))  float  v8f;
typedef __attribute__((ext_vector_type(4)))  int    v4i;

// Pointer-to-AS1-int4 and pointer-to-AS3-int4 (pointee carries the addrspace)
typedef __attribute__((address_space(1))) v4i* gv4i_p;
typedef __attribute__((address_space(3))) v4i* lv4i_p;

#if __has_builtin(__builtin_amdgcn_global_load_async_to_lds_b128)
#define HAS_ASYNC 1
#else
#define HAS_ASYNC 0
#endif

// ---------------------------------------------------------------------------
// Kernel 0: zero the cell accumulator region of the workspace (float4 stores)
// ---------------------------------------------------------------------------
__global__ void __launch_bounds__(256)
zero_f4_kernel(float4* __restrict__ p, int n4) {
    int i = blockIdx.x * blockDim.x + threadIdx.x;
    if (i < n4) p[i] = make_float4(0.f, 0.f, 0.f, 0.f);
}

// ---------------------------------------------------------------------------
// Kernel 1: segment scatter-add  cell[b, ttid[b,s], :] += hidden[b, s, :]
// ---------------------------------------------------------------------------
__global__ void __launch_bounds__(256)
scatter_add_kernel(const float* __restrict__ hidden,
                   const int*   __restrict__ ttid,
                   float*       __restrict__ cell) {
    const int bs = blockIdx.x;              // 0 .. B*S-1
    const int b  = bs >> LOG2_S;
    const int m  = ttid[bs];
    const float4 v = ((const float4*)(hidden + (size_t)bs * H_DIM))[threadIdx.x];
    float* dst = cell + ((size_t)(b * M_DIM + m)) * H_DIM + (threadIdx.x << 2);
    atomicAdd(dst + 0, v.x);
    atomicAdd(dst + 1, v.y);
    atomicAdd(dst + 2, v.z);
    atomicAdd(dst + 3, v.w);
}

// ---------------------------------------------------------------------------
// Kernel 2: split fp32 into bf16 hi/lo planes (done ONCE per operand, so the
// GEMM inner loop carries no conversion VALU work).
// ---------------------------------------------------------------------------
__global__ void __launch_bounds__(256)
split_bf16_kernel(const float* __restrict__ src,
                  __bf16* __restrict__ hi, __bf16* __restrict__ lo, int n4) {
    int i = blockIdx.x * blockDim.x + threadIdx.x;
    if (i >= n4) return;
    float4 v = ((const float4*)src)[i];
    float  x[4] = {v.x, v.y, v.z, v.w};
    v4bf h, l;
#pragma unroll
    for (int j = 0; j < 4; ++j) {
        __bf16 hh = (__bf16)x[j];
        h[j] = hh;
        l[j] = (__bf16)(x[j] - (float)hh);
    }
    ((v4bf*)hi)[i] = h;
    ((v4bf*)lo)[i] = l;
}

// ---------------------------------------------------------------------------
// Kernel 3: WMMA GEMM  act = gelu_tanh(cell @ W^T + b)
// bf16 hi/lo split (3 wmma/tile) with fp32 accumulation ~= fp32 accuracy.
// B (W rows) staged into LDS per-block via async-to-LDS, double buffered.
// ---------------------------------------------------------------------------
__device__ __forceinline__ float gelu_tanh_f(float x) {
    const float c = 0.7978845608028654f;    // sqrt(2/pi)
    float x3 = x * x * x;
    return 0.5f * x * (1.0f + tanhf(c * (x + 0.044715f * x3)));
}

__device__ __forceinline__ void copy16_to_lds(const __bf16* src, __bf16* dst) {
#if HAS_ASYNC
    __builtin_amdgcn_global_load_async_to_lds_b128(
        (gv4i_p)src, (lv4i_p)dst, /*offset=*/0, /*cpol=*/0);
#else
    *(float4*)dst = *(const float4*)src;     // synchronous fallback
#endif
}

__device__ __forceinline__ void wait_async0() {
#if HAS_ASYNC
 #if __has_builtin(__builtin_amdgcn_s_wait_asynccnt)
    __builtin_amdgcn_s_wait_asynccnt(0);
 #else
    asm volatile("s_wait_asynccnt 0" ::: "memory");
 #endif
#endif
}

// Stage one 32-K slab of B (64 rows, hi+lo planes, 8KB) into LDS.
// 128 threads x 4 x b128 (16B) = 8KB.
__device__ __forceinline__ void stage_B(__bf16 (*ldsbuf)[64 * 32],
                                        const __bf16* __restrict__ whi,
                                        const __bf16* __restrict__ wlo,
                                        int n0, int k0, int tid) {
#pragma unroll
    for (int p = 0; p < 2; ++p) {
        const __bf16* plane = p ? wlo : whi;
#pragma unroll
        for (int j = 0; j < 2; ++j) {
            int c    = tid * 2 + j;          // 0..255 chunk id (16B chunks)
            int r    = c >> 2;               // row 0..63
            int koff = (c & 3) * 8;          // 0,8,16,24
            copy16_to_lds(plane + (size_t)(n0 + r) * H_DIM + k0 + koff,
                          &ldsbuf[p][r * 32 + koff]);
        }
    }
}

// A fragment (16x32 bf16 layout): elems 0..7 -> K = k0+half*16+j,
//                                 elems 8..15 -> K = k0+half*16+8+j.
__device__ __forceinline__ v16bf load_a_frag(const __bf16* __restrict__ plane,
                                             size_t rowbase, int k0, int half) {
    v8bf a0 = *(const v8bf*)(plane + rowbase + k0 + half * 16);
    v8bf a1 = *(const v8bf*)(plane + rowbase + k0 + half * 16 + 8);
    v16bf r;
#pragma unroll
    for (int j = 0; j < 8; ++j) { r[j] = a0[j]; r[j + 8] = a1[j]; }
    return r;
}

// B fragment (32x16 bf16 layout): elems 0..15 -> K = half*16 + j, contiguous.
__device__ __forceinline__ v16bf load_b_frag(const __bf16* base, int row, int half) {
    return *(const v16bf*)(base + row * 32 + half * 16);   // 32B aligned
}

#define NT 4   // N-tiles per wave (block: 4 waves x 4 M-tiles, shared N-group)

__global__ void __launch_bounds__(128)
gemm_bias_gelu_kernel(const __bf16* __restrict__ cellhi,
                      const __bf16* __restrict__ celllo,
                      const __bf16* __restrict__ whi,
                      const __bf16* __restrict__ wlo,
                      const float*  __restrict__ bias,
                      float*        __restrict__ act) {
    __shared__ alignas(32) __bf16 ldsB[2][2][64 * 32];   // [buf][plane][row*32+k] = 16KB

    const int tid   = threadIdx.x;
    const int lane  = tid & 31;
    const int wave  = tid >> 5;
    const int mtile = blockIdx.x * 4 + wave;             // 0..127
    const int n0    = blockIdx.y * (16 * NT);            // N-group base
    const int l16   = lane & 15;
    const int half  = lane >> 4;                         // 0 or 1

    const size_t aRow = (size_t)(mtile * 16 + l16) * H_DIM;

    v8f acc[NT] = {};

    stage_B(ldsB[0], whi, wlo, n0, 0, tid);
    int buf = 0;

    for (int k0 = 0; k0 < H_DIM; k0 += 32) {
        wait_async0();          // my slab-(k0) async writes to LDS complete
        __syncthreads();        // everyone's slab ready; prev reads finished
        if (k0 + 32 < H_DIM)
            stage_B(ldsB[buf ^ 1], whi, wlo, n0, k0 + 32, tid);  // overlaps compute

        v16bf ahi = load_a_frag(cellhi, aRow, k0, half);
        v16bf alo = load_a_frag(celllo, aRow, k0, half);

#pragma unroll
        for (int t = 0; t < NT; ++t) {
            v16bf bhi = load_b_frag(ldsB[buf][0], t * 16 + l16, half);
            v16bf blo = load_b_frag(ldsB[buf][1], t * 16 + l16, half);
            acc[t] = __builtin_amdgcn_wmma_f32_16x16x32_bf16(
                false, ahi, false, bhi, (short)0, acc[t], false, false);
            acc[t] = __builtin_amdgcn_wmma_f32_16x16x32_bf16(
                false, alo, false, bhi, (short)0, acc[t], false, false);
            acc[t] = __builtin_amdgcn_wmma_f32_16x16x32_bf16(
                false, ahi, false, blo, (short)0, acc[t], false, false);
        }
        buf ^= 1;
    }

    // Epilogue: C/D layout -> VGPR i holds (M = mtile*16 + half*8 + i, N = n0+t*16+l16)
#pragma unroll
    for (int t = 0; t < NT; ++t) {
        const int   col = n0 + t * 16 + l16;
        const float bv  = bias[col];
#pragma unroll
        for (int i = 0; i < 8; ++i) {
            const int rowD = mtile * 16 + half * 8 + i;
            float x = acc[t][i] + bv;
            act[(size_t)rowD * H_DIM + col] = gelu_tanh_f(x);
        }
    }
}

// ---------------------------------------------------------------------------
// Kernel 4: gather  out[b,s,:] = act[b, ttid[b,s], :]   (coalesced float4)
// ---------------------------------------------------------------------------
__global__ void __launch_bounds__(256)
gather_kernel(const float* __restrict__ act,
              const int*   __restrict__ ttid,
              float*       __restrict__ out) {
    const int bs = blockIdx.x;
    const int b  = bs >> LOG2_S;
    const int m  = ttid[bs];
    const float4* src = (const float4*)(act + ((size_t)(b * M_DIM + m)) * H_DIM);
    float4*       dst = (float4*)(out + (size_t)bs * H_DIM);
    dst[threadIdx.x] = src[threadIdx.x];
}

// ---------------------------------------------------------------------------
extern "C" void kernel_launch(void* const* d_in, const int* in_sizes, int n_in,
                              void* d_out, int out_size, void* d_ws, size_t ws_size,
                              hipStream_t stream) {
    const float* hidden = (const float*)d_in[0];   // [B,S,H] fp32
    const int*   ttid   = (const int*)  d_in[1];   // [B,S] int
    const float* W      = (const float*)d_in[2];   // [H,H] fp32
    const float* bias   = (const float*)d_in[3];   // [H] fp32
    float*       out    = (float*)d_out;           // [B,S,H] fp32

    // Workspace layout (all MB-aligned):
    char*   ws     = (char*)d_ws;
    float*  cell   = (float*)(ws);                             // 8 MB fp32 [R,H]
    float*  act    = (float*)(ws + (size_t)8  * 1024 * 1024);  // 8 MB fp32 [R,H]
    __bf16* cellhi = (__bf16*)(ws + (size_t)16 * 1024 * 1024); // 4 MB
    __bf16* celllo = (__bf16*)(ws + (size_t)20 * 1024 * 1024); // 4 MB
    __bf16* whi    = (__bf16*)(ws + (size_t)24 * 1024 * 1024); // 2 MB
    __bf16* wlo    = (__bf16*)(ws + (size_t)26 * 1024 * 1024); // 2 MB (28 MB total)

    // 0) zero the segment accumulator (ws is poisoned; must zero every call)
    const int n4c = (R_DIM * H_DIM) / 4;           // 524288 float4
    zero_f4_kernel<<<(n4c + 255) / 256, 256, 0, stream>>>((float4*)cell, n4c);

    // 1) segment scatter-add
    scatter_add_kernel<<<B_DIM * S_DIM, 256, 0, stream>>>(hidden, ttid, cell);

    // 2) one-shot bf16 hi/lo splits of cell and W
    split_bf16_kernel<<<(n4c + 255) / 256, 256, 0, stream>>>(cell, cellhi, celllo, n4c);
    const int n4w = (H_DIM * H_DIM) / 4;           // 262144 float4
    split_bf16_kernel<<<(n4w + 255) / 256, 256, 0, stream>>>(W, whi, wlo, n4w);

    // 3) WMMA GEMM + bias + gelu: grid = (128 M-tiles / 4 waves, 1024/64 N-groups)
    gemm_bias_gelu_kernel<<<dim3(32, 16), 128, 0, stream>>>(
        cellhi, celllo, whi, wlo, bias, act);

    // 4) gather back to the sequence (bandwidth-dominant, fully coalesced)
    gather_kernel<<<B_DIM * S_DIM, 256, 0, stream>>>(act, ttid, out);
}
